// SpatialDROW_35940286332968
// MI455X (gfx1250) — matrix operand order repaired and checked by reading, not compiled, not run on hardware
//
#include <hip/hip_runtime.h>
#include <hip/hip_bf16.h>

#ifndef __has_builtin
#define __has_builtin(x) 0
#endif
#if __has_builtin(__builtin_amdgcn_tensor_load_to_lds)
#define HAVE_TDM 1
#else
#define HAVE_TDM 0
#endif
#if defined(__has_include)
#if __has_include(<hip/amd_detail/amd_gfx1250_TDM.h>)
#define TDM_6ARG 1
#endif
#endif
#ifndef TDM_6ARG
#define TDM_6ARG 0
#endif

// ---------------------------------------------------------------------------
// CDNA5 WMMA (wave32): D(16x16 f32) = A(16x32 bf16) x B(32x16 bf16) + C
// ---------------------------------------------------------------------------
typedef __attribute__((ext_vector_type(16))) __bf16 v16bf;
typedef __attribute__((ext_vector_type(8)))  float  v8f;
typedef __attribute__((ext_vector_type(4)))  unsigned int tdm_v4u;
typedef __attribute__((ext_vector_type(8)))  int          tdm_v8i;
typedef __attribute__((ext_vector_type(4)))  int          tdm_v4i;

__device__ __forceinline__ unsigned short f2bf(float f) {
    unsigned int u = __float_as_uint(f);
    unsigned int r = u + 0x7FFFu + ((u >> 16) & 1u);   // round-to-nearest-even
    return (unsigned short)(r >> 16);
}
__device__ __forceinline__ float bf2f(unsigned short h) {
    return __uint_as_float(((unsigned int)h) << 16);
}

#define LRELU(y) ((y) >= 0.0f ? (y) : 0.1f * (y))
#define BN_INV 0.99999500003749969f /* 1/sqrt(1+1e-5) */

#define CCH 256          // channel chunk staged in LDS (64 rows x 256 bf16 = 32 KB)
#define MT  4            // co-tiles per wave (64 output channels)

union BFrag { v16bf v; uint4 q[2]; unsigned short u[16]; };

// ---------------------------------------------------------------------------
// Fused conv1d + bias + BN + LeakyReLU, NHWC activations, as KSZ shifted GEMMs.
// Weights pre-transposed to tap-major bf16: wT[t][co][ci] -- so each (tap, chunk)
// weight stage is a dense 2D tile moved global->LDS by the Tensor Data Mover
// (tensor_load_to_lds, TENSORcnt), with manual bf16 staging as fallback.
// 128 threads = 4 waves/block; wave owns 16 output positions, block owns 64
// output channels; one contiguous-channel B gather (2 x b128) feeds 4 WMMAs.
// ---------------------------------------------------------------------------
template<int KSZ, int PAD>
__global__ void conv_wmma_t(
    const unsigned short* __restrict__ xin, long long rowStrideIn,
    const unsigned short* __restrict__ wT,    // tap-major bf16 weights [t][co][ci]
    const float* __restrict__ bias,
    const float* __restrict__ gam, const float* __restrict__ bet,
    unsigned short* __restrict__ yout,
    int R, int Cin, int Cout, int Pin, int Pout)
{
    extern __shared__ __align__(16) unsigned short wlds[];  // 64 * CCH bf16
    __shared__ float prm[192];                              // bias | gamma | beta
    const int tid  = threadIdx.x;                           // 0..127
    const int lane = tid & 31;
    const int wave = tid >> 5;
    const int half = lane >> 4;
    const int m    = lane & 15;
    const int coBase = blockIdx.y * 64;
    const int tileN  = blockIdx.x * 4 + wave;

    if (tid < 64) {
        prm[tid]       = bias[coBase + tid];
        prm[64 + tid]  = gam[coBase + tid];
        prm[128 + tid] = bet[coBase + tid];
    }

    const long long NT = (long long)R * Pout;
    const int  nCol   = tileN * 16 + m;
    const bool nValid = (nCol < NT);
    const int  row = nValid ? (int)(nCol / Pout) : 0;
    const int  p   = nValid ? (int)(nCol % Pout) : 0;
    const unsigned rowOff = (unsigned)row * (unsigned)rowStrideIn;
    __builtin_prefetch(xin + rowOff, 0, 0);                 // global_prefetch_b8

    const bool vecOK = ((Cin & 7) == 0);                    // b128-aligned channel runs

    v8f acc[MT] = {};
    #pragma unroll
    for (int t = 0; t < KSZ; ++t) {
        const int  ip     = p + t - PAD;
        const bool bvalid = nValid & ((unsigned)ip < (unsigned)Pin);
        const int  ipc    = ip < 0 ? 0 : (ip >= Pin ? Pin - 1 : ip);
        const unsigned aT = rowOff + (unsigned)ipc * (unsigned)Cin + (unsigned)(half * 16);

        for (int cc = 0; cc < Cin; cc += CCH) {
            const int crem = Cin - cc;
            const int cpad = (crem < CCH ? ((crem + 31) & ~31) : CCH);
            const unsigned short* wTt = wT + ((long long)t * Cout + coBase) * Cin + cc;
            __syncthreads();                                // prior chunk consumed
#if HAVE_TDM
            // TDM: DMA 64 x cpad bf16 tile (row stride Cin) global -> LDS.
            // tensor_dim0 = crem so OOB channel reads zero-fill the tile tail.
            if (wave == 0) {
                unsigned long long ga = (unsigned long long)(size_t)wTt;   // byte VA
                unsigned ldsOff = (unsigned)(unsigned long long)(size_t)&wlds[0];
                tdm_v4u g0;
                g0[0] = 1u;                                           // count=1, load D#
                g0[1] = ldsOff;                                       // LDS dest (bytes)
                g0[2] = (unsigned)(ga & 0xFFFFFFFFull);               // global_addr[31:0]
                g0[3] = (unsigned)((ga >> 32) & 0x01FFFFFFull) | (2u << 30); // [56:32]|type=2
                tdm_v8i g1;
                g1[0] = (1 << 16);                                    // data_size = 2 bytes
                g1[1] = (int)(((unsigned)crem & 0xFFFFu) << 16);      // tensor_dim0[15:0]
                g1[2] = (int)((((unsigned)crem >> 16) & 0xFFFFu) | (64u << 16)); // tdim0 hi | tdim1 lo
                g1[3] = (int)(((unsigned)cpad & 0xFFFFu) << 16);      // tile_dim0
                g1[4] = 64;                                           // tile_dim1=64, tile_dim2=0
                g1[5] = Cin;                                          // tensor_dim0_stride[31:0]
                g1[6] = 0;
                g1[7] = 0;
                tdm_v4i gz = {0, 0, 0, 0};
#if TDM_6ARG
                tdm_v8i gz8 = {0, 0, 0, 0, 0, 0, 0, 0};
                __builtin_amdgcn_tensor_load_to_lds(g0, g1, gz, gz, gz8, 0);
#else
                __builtin_amdgcn_tensor_load_to_lds(g0, g1, gz, gz, 0);
#endif
#if __has_builtin(__builtin_amdgcn_s_wait_tensorcnt)
                __builtin_amdgcn_s_wait_tensorcnt(0);
#else
                asm volatile("s_wait_tensorcnt 0x0" ::: "memory");
#endif
            }
#else
            // Fallback: manual bf16 weight staging (contiguous copies)
            for (int mm = wave; mm < 64; mm += 4)
                for (int j = lane; j < cpad; j += 32)
                    wlds[mm * cpad + j] =
                        (j < crem) ? wTt[(long long)mm * Cin + j] : (unsigned short)0;
#endif
            __syncthreads();

            for (int c0 = 0; c0 < cpad; c0 += 32) {
                // B fragment: 16 contiguous channels per lane-half -> 2 x b128
                BFrag Bf;
                const unsigned a = aT + (unsigned)(cc + c0);
                if (vecOK) {
                    const uint4* px = (const uint4*)(xin + a);
                    uint4 q0 = px[0], q1 = px[1];
                    Bf.q[0].x = bvalid ? q0.x : 0u;  Bf.q[0].y = bvalid ? q0.y : 0u;
                    Bf.q[0].z = bvalid ? q0.z : 0u;  Bf.q[0].w = bvalid ? q0.w : 0u;
                    Bf.q[1].x = bvalid ? q1.x : 0u;  Bf.q[1].y = bvalid ? q1.y : 0u;
                    Bf.q[1].z = bvalid ? q1.z : 0u;  Bf.q[1].w = bvalid ? q1.w : 0u;
                } else {                                    // Cin==1 first layer only
                    #pragma unroll
                    for (int e = 0; e < 16; ++e)
                        Bf.u[e] = bvalid ? xin[a + e] : (unsigned short)0;
                }
                // 4 A fragments (16-bit A 16x32 layout) from LDS, 4 WMMAs share B
                #pragma unroll
                for (int mt = 0; mt < MT; ++mt) {
                    const unsigned short* wl = wlds + (mt * 16 + m) * cpad + c0 + half * 16;
                    BFrag A;
                    A.q[0] = *(const uint4*)wl;             // K k0+0..7  (or +16..23)
                    A.q[1] = *(const uint4*)(wl + 8);       // K k0+8..15 (or +24..31)
                    acc[mt] = __builtin_amdgcn_wmma_f32_16x16x32_bf16(
                                  false, A.v, false, Bf.v, (short)0, acc[mt], false, false);
                }
            }
        }
    }

    if (!nValid) return;   // all lanes stayed active through every WMMA
    // Epilogue: C/D layout lane L: N = L%16, VGPR r -> M = (L/16)*8 + r.
    const unsigned oBase = ((unsigned)row * (unsigned)Pout + (unsigned)p) * (unsigned)Cout
                         + (unsigned)(coBase + half * 8);
    #pragma unroll
    for (int mt = 0; mt < MT; ++mt) {
        union { uint4 q; unsigned short u[8]; } st;
        #pragma unroll
        for (int r = 0; r < 8; ++r) {
            int idx = mt * 16 + half * 8 + r;
            float y = (acc[mt][r] + prm[idx]) * (float)BN_INV * prm[64 + idx] + prm[128 + idx];
            y = LRELU(y);
            st.u[r] = f2bf(y);
        }
        *(uint4*)(yout + oBase + mt * 16) = st.q;           // 8 channels, one b128 store
    }
}

// ---------------------------------------------------------------------------
// weight transpose: w (Cout,Cin,KSZ) f32 -> wT (KSZ,Cout,Cin) bf16
// ---------------------------------------------------------------------------
__global__ void transpose_w_kernel(const float* __restrict__ w, unsigned short* __restrict__ wT,
                                   int Cout, int Cin, int KSZ)
{
    int i = blockIdx.x * 256 + threadIdx.x;
    int total = Cout * Cin * KSZ;
    if (i >= total) return;
    int co  = i / (Cin * KSZ);
    int rem = i - co * (Cin * KSZ);
    int ci  = rem / KSZ;
    int t   = rem - ci * KSZ;
    wT[((long long)t * Cout + co) * Cin + ci] = f2bf(w[i]);
}

// ---------------------------------------------------------------------------
// helpers (NHWC bf16)
// ---------------------------------------------------------------------------
__global__ void cvt_f32_bf16(const float* __restrict__ in, unsigned short* __restrict__ out,
                             long long n)
{
    long long i = (long long)blockIdx.x * 256 + threadIdx.x;
    if (i < n) out[i] = f2bf(in[i]);
}

// maxpool2 over position dim, NHWC: grid=(R,Pout), block=C
__global__ void maxpool2_nhwc(const unsigned short* __restrict__ in,
                              unsigned short* __restrict__ out, int C, int Pout)
{
    int row = blockIdx.x, p = blockIdx.y, c = threadIdx.x;
    long long ib = ((long long)row * (2 * Pout) + 2 * p) * C + c;
    long long ob = ((long long)row * Pout + p) * C + c;
    out[ob] = f2bf(fmaxf(bf2f(in[ib]), bf2f(in[ib + C])));
}

// temp0 = feats[:, :, 0]  (bf16 -> f32); layout-agnostic row copy
__global__ void init_temp_kernel(const unsigned short* __restrict__ feats,
                                 float* __restrict__ temp, long long n)
{
    long long i = (long long)blockIdx.x * 256 + threadIdx.x;
    if (i >= n) return;
    long long row = i / 3072, k = i - row * 3072;
    temp[i] = bf2f(feats[(row * 5 + 0) * 3072 + k]);
}

// ---------------------------------------------------------------------------
// Windowed attention gate: one block per (b, n); double-buffered temp.
// ---------------------------------------------------------------------------
__global__ void gate_kernel(const unsigned short* __restrict__ featsSlice, long long rowStrideF,
                            const float* __restrict__ tempIn, float* __restrict__ tempOut,
                            const unsigned short* __restrict__ embF,
                            const unsigned short* __restrict__ embT,
                            float* __restrict__ fused)
{
    __shared__ float simS[7];
    __shared__ float attS[7];
    __shared__ int   colS[7];
    const int blk = blockIdx.x;
    const int b = blk / 450, n = blk - b * 450;
    const int tid = threadIdx.x;

    if (tid < 7) {
        int col = n - 3 + tid;
        col = col < 0 ? 0 : (col > 449 ? 449 : col);
        colS[tid] = col;
        const unsigned short* ef = embF + (long long)blk * 128;
        const unsigned short* et = embT + ((long long)b * 450 + col) * 128;
        float s = 0.0f;
        for (int k = 0; k < 128; ++k) s += bf2f(ef[k]) * bf2f(et[k]);
        simS[tid] = s;
        if (fused) fused[(long long)blk * 7 + tid] = s;  // raw sim window = feat_fused
    }
    __syncthreads();

    if (tid == 0) {
        float mx = -1e30f;
        for (int j = 0; j < 7; ++j) {
            bool uniq = (j == 0) || (colS[j] != colS[j - 1]);  // clipped cols non-decreasing
            attS[j] = uniq ? 1.0f : 0.0f;
            if (uniq && simS[j] > mx) mx = simS[j];
        }
        float sum = 0.0f;
        for (int j = 0; j < 7; ++j) { float e = attS[j] * expf(simS[j] - mx); attS[j] = e; sum += e; }
        float rs = 1.0f / sum;
        for (int j = 0; j < 7; ++j) attS[j] *= rs;
    }
    __syncthreads();

    float att[7]; int col[7];
    #pragma unroll
    for (int j = 0; j < 7; ++j) { att[j] = attS[j]; col[j] = colS[j]; }
    const unsigned short* fr = featsSlice + (long long)blk * rowStrideF;
    for (int k = tid; k < 3072; k += blockDim.x) {
        float ot = 0.0f;
        #pragma unroll
        for (int j = 0; j < 7; ++j)
            ot += att[j] * tempIn[((long long)b * 450 + col[j]) * 3072 + k];
        tempOut[(long long)blk * 3072 + k] = 0.5f * bf2f(fr[k]) + 0.5f * ot;
    }
}

// mean over P=6 positions, NHWC (row, 6, 128): grid=3600, block=128
__global__ void mean_nhwc(const unsigned short* __restrict__ in, float* __restrict__ out)
{
    int row = blockIdx.x, c = threadIdx.x;
    float s = 0.0f;
    #pragma unroll
    for (int p = 0; p < 6; ++p) s += bf2f(in[((long long)row * 6 + p) * 128 + c]);
    out[(long long)row * 128 + c] = s * (1.0f / 6.0f);
}

// 1x1-conv heads: pred_cls (4) and pred_reg (2) from mean features (128)
__global__ void head_kernel(const float* __restrict__ mean,
                            const float* __restrict__ wc, const float* __restrict__ bc,
                            const float* __restrict__ wr, const float* __restrict__ br,
                            float* __restrict__ out)
{
    int r = blockIdx.x * blockDim.x + threadIdx.x;
    if (r >= 3600) return;
    const float* m = mean + (long long)r * 128;
    for (int j = 0; j < 4; ++j) {
        float s = bc[j];
        for (int k = 0; k < 128; ++k) s += wc[j * 128 + k] * m[k];
        out[(long long)r * 4 + j] = s;
    }
    for (int j = 0; j < 2; ++j) {
        float s = br[j];
        for (int k = 0; k < 128; ++k) s += wr[j * 128 + k] * m[k];
        out[14400 + (long long)r * 2 + j] = s;
    }
}

// ---------------------------------------------------------------------------
// Host-side orchestration
// ---------------------------------------------------------------------------
static inline void conv_launch(hipStream_t s, const unsigned short* xin, long long rsIn,
                               const unsigned short* wT,
                               const float* b, const float* g, const float* be,
                               unsigned short* y,
                               int R, int Cin, int Cout, int Pin, int Pout, int ksz)
{
    long long NT = (long long)R * Pout;
    int nTiles = (int)((NT + 15) / 16);
    dim3 grid((unsigned)((nTiles + 3) / 4), (unsigned)(Cout / 64));
    size_t shmem = (size_t)64 * CCH * sizeof(unsigned short);   // 32 KB
    if (ksz == 3)
        conv_wmma_t<3, 1><<<grid, 128, shmem, s>>>(xin, rsIn, wT, b, g, be, y, R, Cin, Cout, Pin, Pout);
    else // ksz == 12, pad 0 (embedding GEMM, K = 3072)
        conv_wmma_t<12, 0><<<grid, 128, shmem, s>>>(xin, rsIn, wT, b, g, be, y, R, Cin, Cout, Pin, Pout);
}

extern "C" void kernel_launch(void* const* d_in, const int* in_sizes, int n_in,
                              void* d_out, int out_size, void* d_ws, size_t ws_size,
                              hipStream_t stream)
{
    (void)in_sizes; (void)n_in; (void)out_size; (void)ws_size;
#define IN(i) ((const float*)d_in[(i)])
    // input order: x, p1[3x(w,b,g,be)], p2[3x4], p3[3x4], p4[2x4], pg[4], wc, bc, wr, br
    const float* x = IN(0);
    const float *wc = IN(49), *bc = IN(50), *wr = IN(51), *br = IN(52);

    // workspace partition (bytes; 4 KB pads make tail-overreads safe)
    char* ws = (char*)d_ws;
    unsigned short* bufA   = (unsigned short*)(ws + 0LL);          // 221.2 MB
    unsigned short* bufB   = (unsigned short*)(ws + 221188096LL);  // 221.2 MB
    unsigned short* feats  = (unsigned short*)(ws + 442376192LL);  // 110.6 MB (18000 rows x 3072)
    float*          tmpA   = (float*)(ws + 552972288LL);           // 44.2 MB
    float*          tmpB   = (float*)(ws + 597213184LL);           // 44.2 MB
    unsigned short* tempbf = (unsigned short*)(ws + 641454080LL);  // 22.1 MB
    unsigned short* embF   = (unsigned short*)(ws + 663576576LL);  // 0.9 MB
    unsigned short* embT   = (unsigned short*)(ws + 664502272LL);  // 0.9 MB
    float*          meanb  = (float*)(ws + 665427968LL);           // 1.8 MB
    unsigned short* xbf    = (unsigned short*)(ws + 667275264LL);  // 1.7 MB
    unsigned short* wTbase = (unsigned short*)(ws + 669007360LL);  // 3.8 MB tap-major weights

    // per-layer tap-major weight offsets (elements, 16-aligned)
    static const long long wtoff[12] = {0, 192, 12480, 37056, 86208, 135360, 233664,
                                        430272, 626880, 1020096, 1413312, 1511616};
    unsigned short* wt[12];
    for (int i = 0; i < 12; ++i) wt[i] = wTbase + wtoff[i];

    float* out = (float*)d_out;
    float* fused = out + 14400 + 7200;   // feat_fused region (8*450*7)

    // ---- one-shot: transpose/convert all conv weights to tap-major bf16 ----
    {
        static const int lw[12]  = {1, 5, 9, 13, 17, 21, 25, 29, 33, 37, 41, 45}; // w input index
        static const int lco[12] = {64, 64, 128, 128, 128, 256, 256, 256, 512, 256, 128, 128};
        static const int lci[12] = {1, 64, 64, 128, 128, 128, 256, 256, 256, 512, 256, 256};
        static const int lks[12] = {3, 3, 3, 3, 3, 3, 3, 3, 3, 3, 3, 12};
        for (int i = 0; i < 12; ++i) {
            int total = lco[i] * lci[i] * lks[i];
            transpose_w_kernel<<<(total + 255) / 256, 256, 0, stream>>>(
                IN(lw[i]), wt[i], lco[i], lci[i], lks[i]);
        }
    }

    // ---- stage A: p1 conv block + pool, p2 conv block + pool (18000 rows, NHWC) ----
    {   long long n = 864000;   // x: (18000, 48, 1) f32 -> bf16 (NHWC == NCHW for Cin=1)
        cvt_f32_bf16<<<(unsigned)((n + 255) / 256), 256, 0, stream>>>(x, xbf, n); }
    conv_launch(stream, xbf,  48,      wt[0], IN(2),  IN(3),  IN(4),  bufA, 18000,   1,  64, 48, 48, 3);
    conv_launch(stream, bufA, 48*64,   wt[1], IN(6),  IN(7),  IN(8),  bufB, 18000,  64,  64, 48, 48, 3);
    conv_launch(stream, bufB, 48*64,   wt[2], IN(10), IN(11), IN(12), bufA, 18000,  64, 128, 48, 48, 3);
    maxpool2_nhwc<<<dim3(18000, 24), 128, 0, stream>>>(bufA, bufB, 128, 24);
    conv_launch(stream, bufB, 24*128,  wt[3], IN(14), IN(15), IN(16), bufA, 18000, 128, 128, 24, 24, 3);
    conv_launch(stream, bufA, 24*128,  wt[4], IN(18), IN(19), IN(20), bufB, 18000, 128, 128, 24, 24, 3);
    conv_launch(stream, bufB, 24*128,  wt[5], IN(22), IN(23), IN(24), bufA, 18000, 128, 256, 24, 24, 3);
    maxpool2_nhwc<<<dim3(18000, 12), 256, 0, stream>>>(bufA, feats, 256, 12);   // feats (18000, 12, 256)

    // ---- stage B: gate loop (S-1 = 4 iterations) ----
    {   long long n = 3600LL * 3072;
        init_temp_kernel<<<(unsigned)((n + 255) / 256), 256, 0, stream>>>(feats, tmpA, n); }
    float* ping = tmpA; float* pong = tmpB;
    for (int i = 1; i < 5; ++i) {
        const unsigned short* fslice = feats + (long long)i * 3072;
        // emb(feat_i): K = 256*12 = 3072 GEMM on WMMA
        conv_launch(stream, fslice, 5LL * 3072, wt[11], IN(46), IN(47), IN(48), embF,
                    3600, 256, 128, 12, 1, 12);
        // emb(temp): convert f32 temp -> bf16, then same GEMM
        {   long long n = 3600LL * 3072;
            cvt_f32_bf16<<<(unsigned)((n + 255) / 256), 256, 0, stream>>>(ping, tempbf, n); }
        conv_launch(stream, tempbf, 3072, wt[11], IN(46), IN(47), IN(48), embT,
                    3600, 256, 128, 12, 1, 12);
        gate_kernel<<<3600, 256, 0, stream>>>(fslice, 5LL * 3072, ping, pong, embF, embT, fused);
        float* t = ping; ping = pong; pong = t;
    }

    // ---- stage C: p3 + pool, p4, mean, heads (3600 rows, NHWC) ----
    {   long long n = 3600LL * 3072;
        cvt_f32_bf16<<<(unsigned)((n + 255) / 256), 256, 0, stream>>>(ping, tempbf, n); }
    conv_launch(stream, tempbf, 3072,   wt[6], IN(26), IN(27), IN(28), bufA, 3600, 256, 256, 12, 12, 3);
    conv_launch(stream, bufA, 12*256,   wt[7], IN(30), IN(31), IN(32), bufB, 3600, 256, 256, 12, 12, 3);
    conv_launch(stream, bufB, 12*256,   wt[8], IN(34), IN(35), IN(36), bufA, 3600, 256, 512, 12, 12, 3);
    maxpool2_nhwc<<<dim3(3600, 6), 512, 0, stream>>>(bufA, bufB, 512, 6);
    conv_launch(stream, bufB, 6*512,    wt[9], IN(38), IN(39), IN(40), bufA, 3600, 512, 256, 6, 6, 3);
    conv_launch(stream, bufA, 6*256,    wt[10], IN(42), IN(43), IN(44), bufB, 3600, 256, 128, 6, 6, 3);
    mean_nhwc<<<3600, 128, 0, stream>>>(bufB, meanb);
    head_kernel<<<(3600 + 255) / 256, 256, 0, stream>>>(meanb, wc, bc, wr, br, out);
#undef IN
}
